// MoE_22436909154698
// MI455X (gfx1250) — matrix operand order repaired and checked by reading, not compiled, not run on hardware
//
#include <hip/hip_runtime.h>

// ---------------- problem constants ----------------
#define B_   4
#define S_   1024
#define D_   1024
#define H_   4096
#define E_   8
#define K_   2
#define BS_  (B_ * S_)      // 4096 tokens
#define MAXN BS_            // worst-case tokens per expert

// ---------------- tiling ----------------
#define TM 128              // token rows per block
#define TN 128              // output cols per block
#define KC 64               // staged contraction chunk = 2 WMMA K-steps

typedef __bf16 bf16_t;
typedef __attribute__((ext_vector_type(16))) __bf16 v16bf;
typedef __attribute__((ext_vector_type(8)))  __bf16 v8bf;
typedef __attribute__((ext_vector_type(8)))  float  v8f;

union V16U { v16bf v; v8bf h[2]; };

__device__ __forceinline__ bf16_t f2bf(float f) { return (bf16_t)f; }

__device__ __forceinline__ unsigned pack2bf(float a, float b) {
    union { bf16_t bv[2]; unsigned u; } p;
    p.bv[0] = (bf16_t)a;
    p.bv[1] = (bf16_t)b;
    return p.u;
}

// A fragment: 16x32 bf16, rows striped in lanes (lanes 0-15: K{0..15},
// lanes 16-31: same M rows, K{16..31}); LDS layout ldsA[M][KC].
__device__ __forceinline__ v16bf load_a_frag(const bf16_t* ldsA, int mBase, int lane) {
    int lrow = lane & 15, lh = lane >> 4;
    const bf16_t* rp = ldsA + (mBase + lrow) * KC;
    V16U u;
    u.h[0] = *(const v8bf*)(rp + lh * 16);
    u.h[1] = *(const v8bf*)(rp + lh * 16 + 8);
    return u.v;
}

// B fragment: 32x16 bf16, lane L<16 holds column N=L, K=0..15; lane L+16 holds
// K=16..31; LDS layout ldsB[N][KC] (K contiguous).
__device__ __forceinline__ v16bf load_b_frag(const bf16_t* ldsB, int nBase, int lane) {
    int lrow = lane & 15, lh = lane >> 4;
    const bf16_t* cp = ldsB + (nBase + lrow) * KC + lh * 16;
    V16U u;
    u.h[0] = *(const v8bf*)(cp);
    u.h[1] = *(const v8bf*)(cp + 8);
    return u.v;
}

// ---------------- init ----------------
__global__ void init_small(int* counts, float* probsum) {
    int t = threadIdx.x;
    if (t < E_) { counts[t] = 0; probsum[t] = 0.f; }
}

// ---------------- gating: wave per token ----------------
__global__ __launch_bounds__(256) void gate_kernel(
    const float* __restrict__ x, const float* __restrict__ noise,
    const float* __restrict__ Wg, const float* __restrict__ bg,
    const float* __restrict__ Wn, const float* __restrict__ bn,
    int* counts, int* idxList, float* wList, float* probsum)
{
    int token = blockIdx.x * 8 + (threadIdx.x >> 5);
    int lane  = threadIdx.x & 31;
    if (token >= BS_) return;

    float ag[E_], an[E_];
#pragma unroll
    for (int e = 0; e < E_; ++e) { ag[e] = 0.f; an[e] = 0.f; }

    const float* xr = x + (size_t)token * D_;
    for (int d = lane; d < D_; d += 32) {
        float xv = xr[d];
#pragma unroll
        for (int e = 0; e < E_; ++e) {
            ag[e] = fmaf(xv, Wg[d * E_ + e], ag[e]);
            an[e] = fmaf(xv, Wn[d * E_ + e], an[e]);
        }
    }
#pragma unroll
    for (int off = 16; off > 0; off >>= 1) {
#pragma unroll
        for (int e = 0; e < E_; ++e) {
            ag[e] += __shfl_xor(ag[e], off);
            an[e] += __shfl_xor(an[e], off);
        }
    }

    if (lane == 0) {
        float logit[E_], nz[E_];
#pragma unroll
        for (int e = 0; e < E_; ++e) {
            logit[e] = ag[e] + bg[e];
            float spi = an[e] + bn[e];
            float sp  = (spi > 20.f) ? spi : log1pf(expf(spi));
            nz[e] = logit[e] + noise[(size_t)token * E_ + e] * sp;
        }
        // top-2 of 8
        int i0 = 0;
#pragma unroll
        for (int e = 1; e < E_; ++e) if (nz[e] > nz[i0]) i0 = e;
        int i1 = (i0 == 0) ? 1 : 0;
#pragma unroll
        for (int e = 0; e < E_; ++e) if (e != i0 && nz[e] > nz[i1]) i1 = e;
        float ex1 = expf(nz[i1] - nz[i0]);
        float inv = 1.f / (1.f + ex1);
        float w0 = inv, w1 = ex1 * inv;

        int p0 = atomicAdd(&counts[i0], 1);
        idxList[i0 * MAXN + p0] = token * 2 + 0;
        wList [i0 * MAXN + p0] = w0;
        int p1 = atomicAdd(&counts[i1], 1);
        idxList[i1 * MAXN + p1] = token * 2 + 1;
        wList [i1 * MAXN + p1] = w1;

        // clean softmax probs for the load-balance loss
        float mx = logit[0];
#pragma unroll
        for (int e = 1; e < E_; ++e) mx = fmaxf(mx, logit[e]);
        float pe[E_], s = 0.f;
#pragma unroll
        for (int e = 0; e < E_; ++e) { pe[e] = expf(logit[e] - mx); s += pe[e]; }
        float is = 1.f / s;
#pragma unroll
        for (int e = 0; e < E_; ++e) atomicAdd(&probsum[e], pe[e] * is);
    }
}

// ---------------- prefix of counts ----------------
__global__ void prefix_kernel(const int* counts, int* basep) {
    if (threadIdx.x == 0 && blockIdx.x == 0) {
        int s = 0;
        for (int e = 0; e < E_; ++e) { basep[e] = s; s += counts[e]; }
    }
}

// ---------------- FFN stage 1: h = gelu(X_e @ W1[e] + b1[e]) ----------------
__global__ __launch_bounds__(256) void ffn1_kernel(
    const float* __restrict__ x, const float* __restrict__ W1,
    const float* __restrict__ b1, const int* __restrict__ counts,
    const int* __restrict__ basep, const int* __restrict__ idxList,
    bf16_t* __restrict__ hbuf)
{
    int e = blockIdx.z;
    int cnt = counts[e];
    int m0 = blockIdx.x * TM;
    if (m0 >= cnt) return;
    int nTile = blockIdx.y;            // H / TN tiles

    __shared__ bf16_t ldsA[TM * KC];   // [M][K]   16 KB
    __shared__ bf16_t ldsB[TN * KC];   // [N][K]   16 KB

    int t = threadIdx.x;
    int lane = t & 31, wave = t >> 5;
    int wm = wave & 1, wn = wave >> 1; // 2 x 4 wave grid
    int lrow = lane & 15, lh = lane >> 4;

    v8f acc[4][2] = {};

    // A-loader role: 32 elements of one gathered x row
    int aRow = t >> 1;
    int aK   = (t & 1) * 32;
    int tok = -1;
    if (m0 + aRow < cnt) tok = idxList[e * MAXN + m0 + aRow] >> 1;
    const float* xrow = (tok >= 0) ? x + (size_t)tok * D_ : x;

    // B-loader role: 8x4 patch of W1, transposed into [N][K]
    int bN0 = (t & 31) * 4;
    int bK0 = (t >> 5) * 8;
    const float* w1p = W1 + (size_t)e * D_ * H_ + (size_t)nTile * TN + bN0;

    for (int k0 = 0; k0 < D_; k0 += KC) {
        // stage A: f32 -> bf16
        {
            unsigned* dst = (unsigned*)(ldsA + aRow * KC + aK);
            if (tok >= 0) {
                const float4* p = (const float4*)(xrow + k0 + aK);
#pragma unroll
                for (int q = 0; q < 8; ++q) {
                    float4 v = p[q];
                    dst[q * 2 + 0] = pack2bf(v.x, v.y);
                    dst[q * 2 + 1] = pack2bf(v.z, v.w);
                }
            } else {
#pragma unroll
                for (int q = 0; q < 16; ++q) dst[q] = 0u;
            }
        }
        // stage B: f32 -> bf16 with transpose
        {
            unsigned* db = (unsigned*)ldsB;
#pragma unroll
            for (int kk = 0; kk < 8; kk += 2) {
                const float* r0 = w1p + (size_t)(k0 + bK0 + kk) * H_;
                const float* r1 = r0 + H_;
                float4 a4 = *(const float4*)r0;
                float4 b4 = *(const float4*)r1;
                db[((bN0 + 0) * KC + bK0 + kk) >> 1] = pack2bf(a4.x, b4.x);
                db[((bN0 + 1) * KC + bK0 + kk) >> 1] = pack2bf(a4.y, b4.y);
                db[((bN0 + 2) * KC + bK0 + kk) >> 1] = pack2bf(a4.z, b4.z);
                db[((bN0 + 3) * KC + bK0 + kk) >> 1] = pack2bf(a4.w, b4.w);
            }
        }
        __syncthreads();

#pragma unroll
        for (int ks = 0; ks < 2; ++ks) {
            const bf16_t* pa = ldsA + ks * 32;
            const bf16_t* pb = ldsB + ks * 32;
            v16bf bf0 = load_b_frag(pb, wn * 32 + 0,  lane);
            v16bf bf1 = load_b_frag(pb, wn * 32 + 16, lane);
#pragma unroll
            for (int i = 0; i < 4; ++i) {
                v16bf af = load_a_frag(pa, wm * 64 + i * 16, lane);
                acc[i][0] = __builtin_amdgcn_wmma_f32_16x16x32_bf16(
                    false, af, false, bf0, (short)0, acc[i][0], false, false);
                acc[i][1] = __builtin_amdgcn_wmma_f32_16x16x32_bf16(
                    false, af, false, bf1, (short)0, acc[i][1], false, false);
            }
        }
        __syncthreads();
    }

    // epilogue: bias + exact GELU -> bf16 h
    int hb = basep[e];
#pragma unroll
    for (int j = 0; j < 2; ++j) {
        int ncol = nTile * TN + wn * 32 + j * 16 + lrow;
        float bias = b1[(size_t)e * H_ + ncol];
#pragma unroll
        for (int i = 0; i < 4; ++i) {
#pragma unroll
            for (int c = 0; c < 8; ++c) {
                int mm = wm * 64 + i * 16 + lh * 8 + c;
                if (m0 + mm < cnt) {
                    float v = acc[i][j][c] + bias;
                    float g = 0.5f * v * (1.f + erff(v * 0.70710678118654752f));
                    hbuf[(size_t)(hb + m0 + mm) * H_ + ncol] = f2bf(g);
                }
            }
        }
    }
}

// ---------------- FFN stage 2: contrib = w * (h @ W2[e] + b2[e]) ----------------
__global__ __launch_bounds__(256) void ffn2_kernel(
    const bf16_t* __restrict__ hbuf, const float* __restrict__ W2,
    const float* __restrict__ b2, const int* __restrict__ counts,
    const int* __restrict__ basep, const int* __restrict__ idxList,
    const float* __restrict__ wList, float* __restrict__ contrib)
{
    int e = blockIdx.z;
    int cnt = counts[e];
    int m0 = blockIdx.x * TM;
    if (m0 >= cnt) return;
    int nTile = blockIdx.y;            // D / TN tiles

    __shared__ bf16_t ldsA[TM * KC];
    __shared__ bf16_t ldsB[TN * KC];

    int t = threadIdx.x;
    int lane = t & 31, wave = t >> 5;
    int wm = wave & 1, wn = wave >> 1;
    int lrow = lane & 15, lh = lane >> 4;

    v8f acc[4][2] = {};

    int hb = basep[e];
    int aRow = t >> 1;
    int aK   = (t & 1) * 32;
    bool aValid = (m0 + aRow < cnt);
    const bf16_t* hrow = hbuf + (size_t)(hb + (aValid ? (m0 + aRow) : 0)) * H_;

    int bN0 = (t & 31) * 4;
    int bK0 = (t >> 5) * 8;
    const float* w2p = W2 + (size_t)e * H_ * D_ + (size_t)nTile * TN + bN0;

    for (int k0 = 0; k0 < H_; k0 += KC) {
        // stage A: already bf16 (64 bytes per thread)
        {
            v8bf* dst = (v8bf*)(ldsA + aRow * KC + aK);
            if (aValid) {
                const v8bf* src = (const v8bf*)(hrow + k0 + aK);
#pragma unroll
                for (int q = 0; q < 4; ++q) dst[q] = src[q];
            } else {
                unsigned* dz = (unsigned*)dst;
#pragma unroll
                for (int q = 0; q < 16; ++q) dz[q] = 0u;
            }
        }
        // stage B: f32 -> bf16 with transpose
        {
            unsigned* db = (unsigned*)ldsB;
#pragma unroll
            for (int kk = 0; kk < 8; kk += 2) {
                const float* r0 = w2p + (size_t)(k0 + bK0 + kk) * D_;
                const float* r1 = r0 + D_;
                float4 a4 = *(const float4*)r0;
                float4 b4 = *(const float4*)r1;
                db[((bN0 + 0) * KC + bK0 + kk) >> 1] = pack2bf(a4.x, b4.x);
                db[((bN0 + 1) * KC + bK0 + kk) >> 1] = pack2bf(a4.y, b4.y);
                db[((bN0 + 2) * KC + bK0 + kk) >> 1] = pack2bf(a4.z, b4.z);
                db[((bN0 + 3) * KC + bK0 + kk) >> 1] = pack2bf(a4.w, b4.w);
            }
        }
        __syncthreads();

#pragma unroll
        for (int ks = 0; ks < 2; ++ks) {
            const bf16_t* pa = ldsA + ks * 32;
            const bf16_t* pb = ldsB + ks * 32;
            v16bf bf0 = load_b_frag(pb, wn * 32 + 0,  lane);
            v16bf bf1 = load_b_frag(pb, wn * 32 + 16, lane);
#pragma unroll
            for (int i = 0; i < 4; ++i) {
                v16bf af = load_a_frag(pa, wm * 64 + i * 16, lane);
                acc[i][0] = __builtin_amdgcn_wmma_f32_16x16x32_bf16(
                    false, af, false, bf0, (short)0, acc[i][0], false, false);
                acc[i][1] = __builtin_amdgcn_wmma_f32_16x16x32_bf16(
                    false, af, false, bf1, (short)0, acc[i][1], false, false);
            }
        }
        __syncthreads();
    }

    // epilogue: scale by gate weight + b2, write disjoint contrib rows
#pragma unroll
    for (int j = 0; j < 2; ++j) {
        int ncol = nTile * TN + wn * 32 + j * 16 + lrow;
        float bias = b2[(size_t)e * D_ + ncol];
#pragma unroll
        for (int i = 0; i < 4; ++i) {
#pragma unroll
            for (int c = 0; c < 8; ++c) {
                int mm = wm * 64 + i * 16 + lh * 8 + c;
                if (m0 + mm < cnt) {
                    int code = idxList[e * MAXN + m0 + mm];  // token*2 + rank
                    float w  = wList [e * MAXN + m0 + mm];
                    contrib[(size_t)code * D_ + ncol] = w * (acc[i][j][c] + bias);
                }
            }
        }
    }
}

// ---------------- combine the two expert contributions per token ----------------
__global__ void combine_kernel(const float* __restrict__ contrib,
                               float* __restrict__ out)
{
    size_t n = (size_t)BS_ * D_;
    for (size_t i = (size_t)blockIdx.x * blockDim.x + threadIdx.x; i < n;
         i += (size_t)gridDim.x * blockDim.x) {
        size_t token = i >> 10;          // / D_ (=1024)
        size_t col   = i & (D_ - 1);
        out[i] = contrib[(token * 2 + 0) * D_ + col] +
                 contrib[(token * 2 + 1) * D_ + col];
    }
}

// ---------------- load-balance loss ----------------
__global__ void loss_kernel(const int* counts, const float* probsum, float* out) {
    if (threadIdx.x == 0 && blockIdx.x == 0) {
        float tot = 0.f;
        for (int e = 0; e < E_; ++e) tot += (float)counts[e];
        float s = 0.f;
        for (int e = 0; e < E_; ++e)
            s += ((float)counts[e] / tot) * (probsum[e] / (float)BS_);
        out[(size_t)BS_ * D_] = s * (float)E_;
    }
}

// ---------------- launch ----------------
extern "C" void kernel_launch(void* const* d_in, const int* in_sizes, int n_in,
                              void* d_out, int out_size, void* d_ws, size_t ws_size,
                              hipStream_t stream) {
    const float* x     = (const float*)d_in[0];
    const float* noise = (const float*)d_in[1];
    const float* Wg    = (const float*)d_in[2];
    const float* bg    = (const float*)d_in[3];
    const float* Wn    = (const float*)d_in[4];
    const float* bn    = (const float*)d_in[5];
    const float* W1    = (const float*)d_in[6];
    const float* b1    = (const float*)d_in[7];
    const float* W2    = (const float*)d_in[8];
    const float* b2    = (const float*)d_in[9];
    float* out = (float*)d_out;

    char* ws = (char*)d_ws;
    int*    counts  = (int*)(ws + 0);
    int*    basep   = (int*)(ws + 64);
    float*  probsum = (float*)(ws + 128);
    int*    idxList = (int*)(ws + 256);
    float*  wListp  = (float*)(ws + 256 + sizeof(int) * E_ * MAXN);
    float*  contrib = (float*)(ws + (1u << 20));
    bf16_t* hbuf    = (bf16_t*)(ws + (1u << 20) +
                                sizeof(float) * (size_t)BS_ * K_ * D_);

    init_small<<<1, 64, 0, stream>>>(counts, probsum);
    gate_kernel<<<BS_ / 8, 256, 0, stream>>>(x, noise, Wg, bg, Wn, bn,
                                             counts, idxList, wListp, probsum);
    prefix_kernel<<<1, 32, 0, stream>>>(counts, basep);
    ffn1_kernel<<<dim3(BS_ / TM, H_ / TN, E_), 256, 0, stream>>>(
        x, W1, b1, counts, basep, idxList, hbuf);
    ffn2_kernel<<<dim3(BS_ / TM, D_ / TN, E_), 256, 0, stream>>>(
        hbuf, W2, b2, counts, basep, idxList, wListp, contrib);
    combine_kernel<<<512, 256, 0, stream>>>(contrib, out);
    loss_kernel<<<1, 32, 0, stream>>>(counts, probsum, out);
}
